// PointNetPlusPlus_88270167868037
// MI455X (gfx1250) — matrix operand order repaired and checked by reading, not compiled, not run on hardware
//
#include <hip/hip_runtime.h>
#include <hip/hip_bf16.h>
#include <hip/hip_fp16.h>

// ---------------------------------------------------------------------------
// PointNet++ forward (batch 0 only -- reference returns out[0]).
// MLP layers run as v_wmma_f32_16x16x32_f16 tiled GEMMs with BN folded into
// f16 weights. FPS / ball-query are scan kernels. Fragment loads are
// explicit 16B vector LDS loads (ds_load_b128) -- the A-fragment per-lane
// data is two contiguous 8-half chunks.
// ---------------------------------------------------------------------------

typedef __attribute__((ext_vector_type(16))) _Float16 v16h;
typedef __attribute__((ext_vector_type(8)))  _Float16 v8h;
typedef __attribute__((ext_vector_type(8)))  float    v8f;

#define NPOINT_   256
#define NSAMPLE_  64
#define RADIUS2_  100.0f
#define EPS_BN    1e-5f

// ---------------------------------------------------------------------------
// WMMA fragment loaders (ISA 7.12.2 layouts, wave32).
// A: 16x32 f16. lane&15 = row M; lane<16 holds K {0..7,16..23}, lane>=16
//    holds K {8..15,24..31}. Per lane: two contiguous 16B chunks.
// B: 32x16 f16 (column n = lane&15); same K split as A: lane<16 holds
//    K {0..7,16..23}, lane>=16 holds K {8..15,24..31} (two 16B chunks).
// D/C: vgpr r, lane l -> row = r + 8*(l/16), col = l%16.
// ---------------------------------------------------------------------------
__device__ __forceinline__ v16h load_a_frag(const _Float16* __restrict__ X,
                                            int ld, int m0, int k0, int lane) {
  const _Float16* row =
      X + (size_t)(m0 + (lane & 15)) * ld + k0 + (((lane >> 4) & 1) << 3);
  v8h lo = *(const v8h*)(row);       // K {0..7}  or {8..15}
  v8h hi = *(const v8h*)(row + 16);  // K {16..23} or {24..31}
  return __builtin_shufflevector(lo, hi, 0, 1, 2, 3, 4, 5, 6, 7, 8, 9, 10, 11,
                                 12, 13, 14, 15);
}

// W stored row-major [C_out][ld]; B(k,n) = W[n0+n][k0+k].
__device__ __forceinline__ v16h load_b_frag(const _Float16* __restrict__ W,
                                            int ld, int n0, int k0, int lane) {
  const _Float16* row =
      W + (size_t)(n0 + (lane & 15)) * ld + k0 + (((lane >> 4) & 1) << 3);
  v8h lo = *(const v8h*)(row);
  v8h hi = *(const v8h*)(row + 16);
  return __builtin_shufflevector(lo, hi, 0, 1, 2, 3, 4, 5, 6, 7, 8, 9, 10, 11,
                                 12, 13, 14, 15);
}

__device__ __forceinline__ v8f wmma_f16(v16h a, v16h b, v8f c) {
  return __builtin_amdgcn_wmma_f32_16x16x32_f16(false, a, false, b,
                                                (short)0, c, false, false);
}

// ---------------------------------------------------------------------------
// Fold BN into weights: W'[o][c] = W[o][c]*g[o]*rsqrt(var+eps) (f16, K-padded)
// bias'[o] = (b[o]-mu[o])*scale + be[o]
// ---------------------------------------------------------------------------
__global__ void prep_weights_kernel(const float* __restrict__ W,
                                    const float* __restrict__ b,
                                    const float* __restrict__ g,
                                    const float* __restrict__ be,
                                    const float* __restrict__ mu,
                                    const float* __restrict__ var,
                                    int cout, int cin, int kpad,
                                    _Float16* __restrict__ Wh,
                                    float* __restrict__ bias) {
  int i = blockIdx.x * blockDim.x + threadIdx.x;
  int total = cout * kpad;
  if (i >= total) return;
  int o = i / kpad, c = i % kpad;
  float A = g[o] * rsqrtf(var[o] + EPS_BN);
  Wh[i] = (c < cin) ? (_Float16)(W[o * cin + c] * A) : (_Float16)0.0f;
  if (c == 0) bias[o] = (b[o] - mu[o]) * A + be[o];
}

// ---------------------------------------------------------------------------
// Farthest point sampling, one block, batch 0. Emits far BEFORE the distance
// update (matches jax.lax.scan output); argmax tie-breaks to lowest index.
// ---------------------------------------------------------------------------
template <int N, int BLK>
__global__ __launch_bounds__(BLK) void fps_kernel(
    const float* __restrict__ xyz, int* __restrict__ fidx,
    float* __restrict__ newxyz) {
  __shared__ float dist[N];
  __shared__ float rv[BLK];
  __shared__ int ri[BLK];
  __shared__ int s_far;
  const int tid = threadIdx.x;
  for (int i = tid; i < N; i += BLK) dist[i] = 1e10f;
  if (tid == 0) s_far = 0;
  __syncthreads();
  for (int it = 0; it < NPOINT_; ++it) {
    const int far = s_far;
    if (tid == 0) {
      fidx[it] = far;
      newxyz[it * 3 + 0] = xyz[far * 3 + 0];
      newxyz[it * 3 + 1] = xyz[far * 3 + 1];
      newxyz[it * 3 + 2] = xyz[far * 3 + 2];
    }
    const float cx = xyz[far * 3 + 0];
    const float cy = xyz[far * 3 + 1];
    const float cz = xyz[far * 3 + 2];
    float bv = -1.0f; int bi = 0;
    for (int i = tid; i < N; i += BLK) {
      float dx = xyz[i * 3 + 0] - cx;
      float dy = xyz[i * 3 + 1] - cy;
      float dz = xyz[i * 3 + 2] - cz;
      float nd = fminf(dist[i], dx * dx + dy * dy + dz * dz);
      dist[i] = nd;
      if (nd > bv) { bv = nd; bi = i; }  // ascending -> lowest index on ties
    }
    rv[tid] = bv; ri[tid] = bi;
    __syncthreads();
    for (int s2 = BLK / 2; s2 > 0; s2 >>= 1) {
      if (tid < s2) {
        float ov = rv[tid + s2]; int oi = ri[tid + s2];
        if (ov > rv[tid] || (ov == rv[tid] && oi < ri[tid])) {
          rv[tid] = ov; ri[tid] = oi;
        }
      }
      __syncthreads();
    }
    if (tid == 0) s_far = ri[0];
    __syncthreads();
  }
}

// ---------------------------------------------------------------------------
// Ball query: one wave per query point; ballot+prefix-popcount compaction
// keeps ascending index order (== sort-then-slice); pad with first hit.
// ---------------------------------------------------------------------------
__global__ __launch_bounds__(32) void ball_kernel(
    const float* __restrict__ xyz, int N, const float* __restrict__ centers,
    int* __restrict__ ball) {
  const int s = blockIdx.x;
  const int lane = threadIdx.x;
  const float cx = centers[s * 3 + 0];
  const float cy = centers[s * 3 + 1];
  const float cz = centers[s * 3 + 2];
  int cnt = 0;
  int firstIdx = 0;
  bool haveFirst = false;
  for (int base = 0; base < N && cnt < NSAMPLE_; base += 32) {
    const int j = base + lane;
    bool pred = false;
    if (j < N) {
      float dx = xyz[j * 3 + 0] - cx;
      float dy = xyz[j * 3 + 1] - cy;
      float dz = xyz[j * 3 + 2] - cz;
      pred = (dx * dx + dy * dy + dz * dz) <= RADIUS2_;
    }
    unsigned mask = (unsigned)__ballot(pred);
    if (!haveFirst && mask != 0u) {
      firstIdx = base + (__ffs(mask) - 1);
      haveFirst = true;
    }
    if (pred) {
      int pos = cnt + __popc(mask & ((1u << lane) - 1u));
      if (pos < NSAMPLE_) ball[s * NSAMPLE_ + pos] = j;
    }
    cnt += __popc(mask);
  }
  for (int p = cnt + lane; p < NSAMPLE_; p += 32)
    ball[s * NSAMPLE_ + p] = firstIdx;
}

// ---------------------------------------------------------------------------
// SA shared-MLP via WMMA.  ROWS rows (samples) per group, processed in
// 64-row chunks; 128 threads = 4 waves, each wave owns one 16-row M-tile.
//   layer1: X(64 x K1PAD) @ W1'(64 x K1PAD)^T -> bias+relu -> Y1 (LDS f16)
//   layer2: Y1(64 x 64)   @ W2'(128 x 64)^T   -> bias+relu -> column max
// Column-max accumulated in LDS via int atomicMax (relu output >= 0).
// MODE 0: sa1 (3 xyz-diff channels, zero-padded to 32)
// MODE 1: sa2 (3 xyz-diff + first 125 feature channels -> K=128)
// MODE 2: sa3 (no gather: [l2x | l2p] 131 channels padded to 160, 256 rows)
// ---------------------------------------------------------------------------
template <int ROWS, int K1PAD, int MODE>
__global__ __launch_bounds__(128) void sa_mlp_kernel(
    const float* __restrict__ xyz,      // Nx3 source pts (M0/1) or l2x (M2)
    const float* __restrict__ feat,     // Nx128 feats (M1) or l2p (M2)
    const float* __restrict__ centers,  // NPOINTx3 (M0/1)
    const int* __restrict__ ball,       // NPOINTx64 (M0/1)
    const _Float16* __restrict__ W1h, const float* __restrict__ B1,
    const _Float16* __restrict__ W2h, const float* __restrict__ B2,
    float* __restrict__ out) {          // NPOINTx128 (M0/1) or 128 (M2)
  constexpr int WBUF = (64 * K1PAD > 128 * 64) ? 64 * K1PAD : 128 * 64;
  constexpr int NCHUNK = ROWS / 64;
  constexpr int KT1 = K1PAD / 32;
  __shared__ alignas(16) _Float16 Xs[64 * K1PAD];
  __shared__ alignas(16) _Float16 Y1s[64 * 64];
  __shared__ alignas(16) _Float16 Ws[WBUF];
  __shared__ float b1s[64];
  __shared__ float b2s[128];
  __shared__ int colmax[128];

  const int tid = threadIdx.x;
  const int lane = tid & 31;
  const int wave = tid >> 5;  // M-tile id, uniform per wave
  const int s = blockIdx.x;

  if (tid < 128) colmax[tid] = 0;
  if (tid < 64) b1s[tid] = B1[tid];
  if (tid < 128) b2s[tid] = B2[tid];

  float ccx = 0.f, ccy = 0.f, ccz = 0.f;
  if (MODE != 2) {
    ccx = centers[s * 3 + 0];
    ccy = centers[s * 3 + 1];
    ccz = centers[s * 3 + 2];
  }

  for (int chunk = 0; chunk < NCHUNK; ++chunk) {
    // ---- stage X (2 threads per row) ----
    {
      const int r = tid >> 1, h = tid & 1;
      if (MODE == 2) {
        const int pr = chunk * 64 + r;
        if (h == 0) {
          Xs[r * K1PAD + 0] = (_Float16)xyz[pr * 3 + 0];
          Xs[r * K1PAD + 1] = (_Float16)xyz[pr * 3 + 1];
          Xs[r * K1PAD + 2] = (_Float16)xyz[pr * 3 + 2];
        }
        for (int c = 3 + h; c < K1PAD; c += 2)
          Xs[r * K1PAD + c] =
              (c < 131) ? (_Float16)feat[pr * 128 + (c - 3)] : (_Float16)0.0f;
      } else {
        const int nb = ball[s * NSAMPLE_ + r];
        if (h == 0) {
          Xs[r * K1PAD + 0] = (_Float16)(xyz[nb * 3 + 0] - ccx);
          Xs[r * K1PAD + 1] = (_Float16)(xyz[nb * 3 + 1] - ccy);
          Xs[r * K1PAD + 2] = (_Float16)(xyz[nb * 3 + 2] - ccz);
        }
        if (MODE == 0) {
          for (int c = 3 + h; c < K1PAD; c += 2)
            Xs[r * K1PAD + c] = (_Float16)0.0f;
        } else {  // MODE 1: channels 3..127 <- feat[nb][0..124]
          for (int c = 3 + h; c < K1PAD; c += 2)
            Xs[r * K1PAD + c] = (_Float16)feat[nb * 128 + (c - 3)];
        }
      }
    }
    // ---- stage W1' (16B vector copies) ----
    for (int i = tid * 8; i < 64 * K1PAD; i += 128 * 8)
      *(v8h*)(Ws + i) = *(const v8h*)(W1h + i);
    __builtin_prefetch(W2h, 0, 1);  // global_prefetch for layer-2 weights
    __syncthreads();

    // ---- layer 1: 16x64 strip per wave ----
    v16h afr[KT1];
#pragma unroll
    for (int kt = 0; kt < KT1; ++kt)
      afr[kt] = load_a_frag(Xs, K1PAD, wave * 16, kt * 32, lane);
#pragma unroll
    for (int nt = 0; nt < 4; ++nt) {
      v8f acc = {};
#pragma unroll
      for (int kt = 0; kt < KT1; ++kt) {
        v16h bf = load_b_frag(Ws, K1PAD, nt * 16, kt * 32, lane);
        acc = wmma_f16(afr[kt], bf, acc);
      }
      const int col = nt * 16 + (lane & 15);
      const float bb = b1s[col];
      const int rbase = wave * 16 + 8 * (lane >> 4);
#pragma unroll
      for (int r = 0; r < 8; ++r) {
        float v = acc[r] + bb;
        v = v > 0.f ? v : 0.f;
        Y1s[(rbase + r) * 64 + col] = (_Float16)v;
      }
    }
    __syncthreads();

    // ---- stage W2' (reuse Ws, 16B vector copies) ----
    for (int i = tid * 8; i < 128 * 64; i += 128 * 8)
      *(v8h*)(Ws + i) = *(const v8h*)(W2h + i);
    __syncthreads();

    // ---- layer 2: 16x128 strip per wave, fused column max ----
    v16h a20 = load_a_frag(Y1s, 64, wave * 16, 0, lane);
    v16h a21 = load_a_frag(Y1s, 64, wave * 16, 32, lane);
#pragma unroll
    for (int nt = 0; nt < 8; ++nt) {
      v8f acc = {};
      acc = wmma_f16(a20, load_b_frag(Ws, 64, nt * 16, 0, lane), acc);
      acc = wmma_f16(a21, load_b_frag(Ws, 64, nt * 16, 32, lane), acc);
      const int col = nt * 16 + (lane & 15);
      const float bb = b2s[col];
      float m = 0.f;  // relu output >= 0
#pragma unroll
      for (int r = 0; r < 8; ++r) {
        float v = acc[r] + bb;
        v = v > 0.f ? v : 0.f;
        m = v > m ? v : m;
      }
      m = fmaxf(m, __shfl_xor(m, 16, 32));
      if (lane < 16) atomicMax(&colmax[col], __float_as_int(m));
    }
    __syncthreads();
  }

  if (tid < 128) {
    float v = __int_as_float(colmax[tid]);
    if (MODE == 2) out[tid] = v;
    else out[s * 128 + tid] = v;
  }
}

// ---------------------------------------------------------------------------
// Final head: h1 = relu(x @ W1^T + b1); out = softmax(h1 @ W2^T + b2)
// ---------------------------------------------------------------------------
__global__ __launch_bounds__(256) void head_kernel(
    const float* __restrict__ x, const float* __restrict__ W1,
    const float* __restrict__ b1, const float* __restrict__ W2,
    const float* __restrict__ b2, float* __restrict__ out) {
  __shared__ float h1[256];
  __shared__ float logit[10];
  const int t = threadIdx.x;
  float acc = b1[t];
  for (int c = 0; c < 128; ++c) acc += x[c] * W1[t * 128 + c];
  h1[t] = acc > 0.f ? acc : 0.f;
  __syncthreads();
  if (t < 10) {
    float a = b2[t];
    for (int o = 0; o < 256; ++o) a += h1[o] * W2[t * 256 + o];
    logit[t] = a;
  }
  __syncthreads();
  if (t == 0) {
    float mx = logit[0];
    for (int k = 1; k < 10; ++k) mx = fmaxf(mx, logit[k]);
    float ssum = 0.f, e[10];
    for (int k = 0; k < 10; ++k) { e[k] = __expf(logit[k] - mx); ssum += e[k]; }
    for (int k = 0; k < 10; ++k) out[k] = e[k] / ssum;
  }
}

// ---------------------------------------------------------------------------
// Launch. Input order assumed: insertion order of setup_inputs() dicts:
//   0: xyz (16*8192*3)
//   sa1 L0: 1..6  (W,b,g,be,mu,var)   sa1 L1: 7..12
//   sa2 L0: 13..18                    sa2 L1: 19..24
//   sa3 L0: 25..30                    sa3 L1: 31..36
//   37: W1 (256x128)  38: b1  39: W2 (10x256)  40: b2
// ---------------------------------------------------------------------------
extern "C" void kernel_launch(void* const* d_in, const int* in_sizes, int n_in,
                              void* d_out, int out_size, void* d_ws,
                              size_t ws_size, hipStream_t stream) {
  (void)in_sizes; (void)n_in; (void)out_size; (void)ws_size;
  const float* xyz0 = (const float*)d_in[0];  // batch 0 = first 8192*3 floats

  auto Lf = [&](int i) { return (const float*)d_in[i]; };

  // workspace layout (256B aligned)
  char* ws = (char*)d_ws;
  size_t off = 0;
  auto take = [&](size_t bytes) {
    char* p = ws + off;
    off = (off + bytes + 255) & ~(size_t)255;
    return p;
  };
  _Float16* sa1W1h = (_Float16*)take(64 * 32 * 2);
  float*    sa1b1  = (float*)take(64 * 4);
  _Float16* sa1W2h = (_Float16*)take(128 * 64 * 2);
  float*    sa1b2  = (float*)take(128 * 4);
  _Float16* sa2W1h = (_Float16*)take(64 * 128 * 2);
  float*    sa2b1  = (float*)take(64 * 4);
  _Float16* sa2W2h = (_Float16*)take(128 * 64 * 2);
  float*    sa2b2  = (float*)take(128 * 4);
  _Float16* sa3W1h = (_Float16*)take(64 * 160 * 2);
  float*    sa3b1  = (float*)take(64 * 4);
  _Float16* sa3W2h = (_Float16*)take(128 * 64 * 2);
  float*    sa3b2  = (float*)take(128 * 4);
  int*   fidx1   = (int*)take(NPOINT_ * 4);
  float* newxyz1 = (float*)take(NPOINT_ * 3 * 4);
  int*   ball1   = (int*)take(NPOINT_ * NSAMPLE_ * 4);
  float* l1p     = (float*)take(NPOINT_ * 128 * 4);
  int*   fidx2   = (int*)take(NPOINT_ * 4);
  float* newxyz2 = (float*)take(NPOINT_ * 3 * 4);  // == l2x
  int*   ball2   = (int*)take(NPOINT_ * NSAMPLE_ * 4);
  float* l2p     = (float*)take(NPOINT_ * 128 * 4);
  float* x3      = (float*)take(128 * 4);

  // --- fold BN + convert weights ---
  auto prep = [&](int base, int cout, int cin, int kpad, _Float16* Wh,
                  float* bias) {
    int total = cout * kpad;
    prep_weights_kernel<<<(total + 255) / 256, 256, 0, stream>>>(
        Lf(base + 0), Lf(base + 1), Lf(base + 2), Lf(base + 3), Lf(base + 4),
        Lf(base + 5), cout, cin, kpad, Wh, bias);
  };
  prep(1, 64, 3, 32, sa1W1h, sa1b1);
  prep(7, 128, 64, 64, sa1W2h, sa1b2);
  prep(13, 64, 128, 128, sa2W1h, sa2b1);
  prep(19, 128, 64, 64, sa2W2h, sa2b2);
  prep(25, 64, 131, 160, sa3W1h, sa3b1);
  prep(31, 128, 64, 64, sa3W2h, sa3b2);

  // --- SA1 ---
  fps_kernel<8192, 1024><<<1, 1024, 0, stream>>>(xyz0, fidx1, newxyz1);
  ball_kernel<<<NPOINT_, 32, 0, stream>>>(xyz0, 8192, newxyz1, ball1);
  sa_mlp_kernel<64, 32, 0><<<NPOINT_, 128, 0, stream>>>(
      xyz0, nullptr, newxyz1, ball1, sa1W1h, sa1b1, sa1W2h, sa1b2, l1p);

  // --- SA2 ---
  fps_kernel<256, 256><<<1, 256, 0, stream>>>(newxyz1, fidx2, newxyz2);
  ball_kernel<<<NPOINT_, 32, 0, stream>>>(newxyz1, 256, newxyz2, ball2);
  sa_mlp_kernel<64, 128, 1><<<NPOINT_, 128, 0, stream>>>(
      newxyz1, l1p, newxyz2, ball2, sa2W1h, sa2b1, sa2W2h, sa2b2, l2p);

  // --- SA3 (global, 256 rows, 1 block) ---
  sa_mlp_kernel<256, 160, 2><<<1, 128, 0, stream>>>(
      newxyz2, l2p, nullptr, nullptr, sa3W1h, sa3b1, sa3W2h, sa3b2, x3);

  // --- head ---
  head_kernel<<<1, 256, 0, stream>>>(x3, Lf(37), Lf(38), Lf(39), Lf(40),
                                     (float*)d_out);
}
